// MultiHeadAttention_43413529428297
// MI455X (gfx1250) — matrix operand order repaired and checked by reading, not compiled, not run on hardware
//
#include <hip/hip_runtime.h>
#include <hip/hip_bf16.h>

#define D_MODEL 1024
#define NHEADS  16
#define DHEAD   64
#define BATCH   4
#define SEQ     2048
#define MTOK    (BATCH * SEQ)   // 8192 tokens

typedef __attribute__((ext_vector_type(16))) _Float16 v16h;
typedef __attribute__((ext_vector_type(8)))  _Float16 v8h;
typedef __attribute__((ext_vector_type(8)))  float    v8f;

__device__ __forceinline__ v8f zero8() {
    v8f z;
#pragma unroll
    for (int i = 0; i < 8; ++i) z[i] = 0.0f;
    return z;
}

__device__ __forceinline__ v8f wmma_f16(v16h a, v16h b, v8f c) {
    // D = A(16x32 f16) * B(32x16 f16) + C(16x16 f32)
    return __builtin_amdgcn_wmma_f32_16x16x32_f16(false, a, false, b, (short)0, c, false, false);
}

// A-fragment: 16x32 tile, row-major [M][K], ld in halves (rows 16B aligned).
// ISA 7.12.2: lanes 0-15 M=0..15 hold K {0..7,16..23}; lanes 16-31 hold K {8..15,24..31}.
__device__ __forceinline__ v16h load_a(const _Float16* base, int ld) {
    const int lane = threadIdx.x & 31;
    const _Float16* r = base + (lane & 15) * ld + ((lane >> 4) << 3);
    v8h lo = *(const v8h*)r;
    v8h hi = *(const v8h*)(r + 16);
    v16h a;
#pragma unroll
    for (int i = 0; i < 8; ++i) { a[i] = lo[i]; a[i + 8] = hi[i]; }
    return a;
}

// B-fragment from B^T layout [N][K] row-major: lane n = lane&15, K = (lane>>4)*8 + {0..7,16..23}.
__device__ __forceinline__ v16h load_b(const _Float16* base, int ld) {
    const int lane = threadIdx.x & 31;
    const _Float16* r = base + (lane & 15) * ld + ((lane >> 4) << 3);
    v8h lo = *(const v8h*)r;
    v8h hi = *(const v8h*)(r + 16);
    v16h b;
#pragma unroll
    for (int i = 0; i < 8; ++i) { b[i] = lo[i]; b[i + 8] = hi[i]; }
    return b;
}

// ---------------- CDNA5 async global->LDS staging (ASYNCcnt path) ----------------

__device__ __forceinline__ unsigned int lds_addr32(const void* p) {
    // Aperture decode: LDS flat address low 32 bits == LDS byte address.
    return (unsigned int)(unsigned long long)p;
}

// GVS mode: mem = SADDR(u64, uniform) + VADDR(i32, per-lane) ; LDS dest from VDST vgpr.
__device__ __forceinline__ void async_ld_b128(unsigned int ldsa, const void* gbase,
                                              unsigned int gof) {
    asm volatile("global_load_async_to_lds_b128 %0, %1, %2"
                 :: "v"(ldsa), "v"(gof), "s"(gbase)
                 : "memory");
}

__device__ __forceinline__ void wait_async0() {
    asm volatile("s_wait_asynccnt 0x0" ::: "memory");
}

// ---------------- conversion kernels ----------------

__global__ void cvt_f16_kernel(const float* __restrict__ in, _Float16* __restrict__ out, int n) {
    int i = blockIdx.x * blockDim.x + threadIdx.x;
    if (i < n) out[i] = (_Float16)in[i];
}

// Wt[n][k] = (f16) W[k][n]   (1024x1024)
__global__ __launch_bounds__(256) void wt_cvt_kernel(const float* __restrict__ W,
                                                     _Float16* __restrict__ Wt) {
    __shared__ float t[32][33];
    const int tx = threadIdx.x, ty = threadIdx.y;       // 32 x 8
    const int k0 = blockIdx.y * 32, n0 = blockIdx.x * 32;
#pragma unroll
    for (int j = 0; j < 32; j += 8)
        t[ty + j][tx] = W[(size_t)(k0 + ty + j) * D_MODEL + n0 + tx];
    __syncthreads();
#pragma unroll
    for (int j = 0; j < 32; j += 8)
        Wt[(size_t)(n0 + ty + j) * D_MODEL + k0 + tx] = (_Float16)t[tx][ty + j];
}

// ---------------- QKV projection GEMM (async-staged, double-buffered) ----------------
//   Q: [b][h][s][d] (scaled by 1/8), K: [b][h][s][d], V: [b][h][d][s] (transposed)

#define GT_M 128
#define GT_N 128
#define GT_K 32
#define NKSTEP (D_MODEL / GT_K)   // 32
#define LDS_LD (GT_K + 8)         // 40 halves: 16B aligned, bank-spread

__global__ __launch_bounds__(256) void qkv_gemm_kernel(
    const _Float16* __restrict__ xh,
    const _Float16* __restrict__ Wtq, const _Float16* __restrict__ Wtk,
    const _Float16* __restrict__ Wtv,
    const float* __restrict__ bq, const float* __restrict__ bk, const float* __restrict__ bv,
    _Float16* __restrict__ Q, _Float16* __restrict__ K, _Float16* __restrict__ Vt) {
    __shared__ _Float16 sA[2][GT_M * LDS_LD];
    __shared__ _Float16 sB[2][GT_N * LDS_LD];

    const int proj = blockIdx.z;
    const _Float16* Wt = (proj == 0) ? Wtq : ((proj == 1) ? Wtk : Wtv);
    const float* bias  = (proj == 0) ? bq  : ((proj == 1) ? bk  : bv);

    const int m0 = blockIdx.x * GT_M;
    const int n0 = blockIdx.y * GT_N;
    const int tid = threadIdx.x;
    const int wave = tid >> 5;
    const int wm = wave & 1;   // 2 wave-rows of 64
    const int wn = wave >> 1;  // 4 wave-cols of 32

    v8f acc[4][2];
#pragma unroll
    for (int i = 0; i < 4; ++i)
#pragma unroll
        for (int j = 0; j < 2; ++j) acc[i][j] = zero8();

    const int lr = tid >> 1;            // 0..127 : tile row
    const int lc = (tid & 1) << 4;      // 0 / 16 : half-row of 16 halves

    auto stage = [&](int buf, int k0) {
        unsigned int la = lds_addr32(&sA[buf][lr * LDS_LD + lc]);
        unsigned int ga = (unsigned int)(((m0 + lr) * D_MODEL + k0 + lc) * 2);
        async_ld_b128(la, xh, ga);
        async_ld_b128(la + 16, xh, ga + 16);
        unsigned int lb = lds_addr32(&sB[buf][lr * LDS_LD + lc]);
        unsigned int gb = (unsigned int)(((n0 + lr) * D_MODEL + k0 + lc) * 2);
        async_ld_b128(lb, Wt, gb);
        async_ld_b128(lb + 16, Wt, gb + 16);
    };

    stage(0, 0);
    wait_async0();
    __syncthreads();

    int buf = 0;
    for (int kt = 0; kt < NKSTEP; ++kt) {
        if (kt + 1 < NKSTEP) stage(buf ^ 1, (kt + 1) * GT_K);  // prefetch next tile

        v16h af[4], bf[2];
#pragma unroll
        for (int i = 0; i < 4; ++i)
            af[i] = load_a(&sA[buf][(wm * 64 + i * 16) * LDS_LD], LDS_LD);
#pragma unroll
        for (int j = 0; j < 2; ++j)
            bf[j] = load_b(&sB[buf][(wn * 32 + j * 16) * LDS_LD], LDS_LD);
#pragma unroll
        for (int i = 0; i < 4; ++i)
#pragma unroll
            for (int j = 0; j < 2; ++j)
                acc[i][j] = wmma_f16(af[i], bf[j], acc[i][j]);

        if (kt + 1 < NKSTEP) wait_async0();  // next tile landed (own wave)
        __syncthreads();                     // visible to all waves; readers done
        buf ^= 1;
    }

    const int lane = tid & 31;
    const int rowh = (lane >> 4) * 8;   // C layout: VGPR e -> row e or 8+e
    const int coln = lane & 15;
#pragma unroll
    for (int i = 0; i < 4; ++i)
#pragma unroll
        for (int j = 0; j < 2; ++j)
#pragma unroll
            for (int e = 0; e < 8; ++e) {
                int m = m0 + wm * 64 + i * 16 + rowh + e;
                int n = n0 + wn * 32 + j * 16 + coln;
                float v = acc[i][j][e] + bias[n];
                int b = m >> 11, s = m & (SEQ - 1);
                int h = n >> 6, d = n & (DHEAD - 1);
                if (proj == 0) {
                    Q[(((size_t)b * NHEADS + h) * SEQ + s) * DHEAD + d] =
                        (_Float16)(v * 0.125f);  // fold 1/sqrt(64)
                } else if (proj == 1) {
                    K[(((size_t)b * NHEADS + h) * SEQ + s) * DHEAD + d] = (_Float16)v;
                } else {
                    Vt[(((size_t)b * NHEADS + h) * DHEAD + d) * SEQ + s] = (_Float16)v;
                }
            }
}

// ---------------- flash attention ----------------
// Block = 128 queries (8 waves x 16 rows), 64-key tiles, online softmax.

#define QTILE 128
#define KTILE 64
#define ALD   72   // 64 + 8 pad halves

__global__ __launch_bounds__(256) void attn_kernel(
    const _Float16* __restrict__ Q, const _Float16* __restrict__ K,
    const _Float16* __restrict__ Vt, _Float16* __restrict__ ctx) {
    __shared__ _Float16 sK[KTILE * ALD];       // keys row-major [k][d]  (== B^T of K^T)
    __shared__ _Float16 sV[DHEAD * ALD];       // V^T  [d][k]            (== B^T of V)
    __shared__ _Float16 sP[8 * 16 * ALD];      // per-wave P staging (C->A relayout)

    const int bh = blockIdx.y;                 // b*16 + h
    const int q0 = blockIdx.x * QTILE;
    const int tid = threadIdx.x;
    const int wave = tid >> 5, lane = tid & 31;
    const int qb = q0 + wave * 16;
    const size_t hbase = (size_t)bh * SEQ * DHEAD;
    const _Float16* Kbase = K + hbase;
    const _Float16* Vbase = Vt + hbase;

    v16h aq[2];
#pragma unroll
    for (int t = 0; t < 2; ++t)
        aq[t] = load_a(Q + hbase + (size_t)qb * DHEAD + t * 32, DHEAD);

    v8f o[4];
#pragma unroll
    for (int i = 0; i < 4; ++i) o[i] = zero8();
    float m8[8], l8[8];
#pragma unroll
    for (int e = 0; e < 8; ++e) { m8[e] = -1e30f; l8[e] = 0.0f; }

    _Float16* sPw = sP + wave * 16 * ALD;
    const int rowh = (lane >> 4) * 8;
    const int coln = lane & 15;

    const int ntiles = (q0 + QTILE) / KTILE;   // causal bound for this block
    for (int jt = 0; jt < ntiles; ++jt) {
        const int j0 = jt * KTILE;
        __syncthreads();                       // previous tile's readers done
        {   // async-stage K tile [64 keys][64 d] and V^T tile [64 d][64 keys]
            int r = tid >> 2, cs = (tid & 3) << 4;
            unsigned int lk = lds_addr32(&sK[r * ALD + cs]);
            unsigned int gk = (unsigned int)(((j0 + r) * DHEAD + cs) * 2);
            async_ld_b128(lk, Kbase, gk);
            async_ld_b128(lk + 16, Kbase, gk + 16);
            unsigned int lv = lds_addr32(&sV[r * ALD + cs]);
            unsigned int gv = (unsigned int)((r * SEQ + j0 + cs) * 2);
            async_ld_b128(lv, Vbase, gv);
            async_ld_b128(lv + 16, Vbase, gv + 16);
        }
        wait_async0();
        __syncthreads();

        // scores S = Q * K^T  (Q pre-scaled)
        v8f sc[4];
#pragma unroll
        for (int nf = 0; nf < 4; ++nf) {
            v8f z = zero8();
            z = wmma_f16(aq[0], load_b(&sK[nf * 16 * ALD], ALD), z);
            z = wmma_f16(aq[1], load_b(&sK[nf * 16 * ALD + 32], ALD), z);
            sc[nf] = z;
        }
        // causal mask
#pragma unroll
        for (int nf = 0; nf < 4; ++nf)
#pragma unroll
            for (int e = 0; e < 8; ++e) {
                int qr = qb + rowh + e;
                int kc = j0 + nf * 16 + coln;
                if (kc > qr) sc[nf][e] = -1e30f;
            }
        // online softmax per query row (row lives on 16 lanes of one half)
#pragma unroll
        for (int e = 0; e < 8; ++e) {
            float mx = fmaxf(fmaxf(sc[0][e], sc[1][e]), fmaxf(sc[2][e], sc[3][e]));
#pragma unroll
            for (int msk = 1; msk < 16; msk <<= 1)
                mx = fmaxf(mx, __shfl_xor(mx, msk, 32));
            float mnew = fmaxf(m8[e], mx);
            float corr = __expf(m8[e] - mnew);
            float rs = 0.0f;
#pragma unroll
            for (int nf = 0; nf < 4; ++nf) {
                float p = __expf(sc[nf][e] - mnew);
                sc[nf][e] = p;
                rs += p;
            }
#pragma unroll
            for (int msk = 1; msk < 16; msk <<= 1)
                rs += __shfl_xor(rs, msk, 32);
            l8[e] = l8[e] * corr + rs;
            m8[e] = mnew;
#pragma unroll
            for (int df = 0; df < 4; ++df) o[df][e] *= corr;
        }
        // C-layout -> A-layout relayout of P through per-wave LDS
#pragma unroll
        for (int nf = 0; nf < 4; ++nf)
#pragma unroll
            for (int e = 0; e < 8; ++e)
                sPw[(rowh + e) * ALD + nf * 16 + coln] = (_Float16)sc[nf][e];
        asm volatile("s_wait_dscnt 0x0" ::: "memory");  // order per-wave LDS store->load
        v16h ap0 = load_a(sPw, ALD);
        v16h ap1 = load_a(sPw + 32, ALD);
        // O += P * V   (B^T = V^T tile in sV)
#pragma unroll
        for (int df = 0; df < 4; ++df) {
            o[df] = wmma_f16(ap0, load_b(&sV[df * 16 * ALD], ALD), o[df]);
            o[df] = wmma_f16(ap1, load_b(&sV[df * 16 * ALD + 32], ALD), o[df]);
        }
    }

    // write merged-heads ctx [token][h*64+d] as f16
    const int b = bh >> 4, h = bh & 15;
#pragma unroll
    for (int df = 0; df < 4; ++df)
#pragma unroll
        for (int e = 0; e < 8; ++e) {
            int s = qb + rowh + e;
            size_t tok = (size_t)b * SEQ + s;
            int col = h * DHEAD + df * 16 + coln;
            ctx[tok * D_MODEL + col] = (_Float16)(o[df][e] / l8[e]);
        }
}

// ---------------- output projection GEMM (async-staged, double-buffered) ----------------

__global__ __launch_bounds__(256) void out_gemm_kernel(
    const _Float16* __restrict__ ctxh, const _Float16* __restrict__ Wto,
    const float* __restrict__ bo, float* __restrict__ out) {
    __shared__ _Float16 sA[2][GT_M * LDS_LD];
    __shared__ _Float16 sB[2][GT_N * LDS_LD];

    const int m0 = blockIdx.x * GT_M;
    const int n0 = blockIdx.y * GT_N;
    const int tid = threadIdx.x;
    const int wave = tid >> 5;
    const int wm = wave & 1, wn = wave >> 1;

    v8f acc[4][2];
#pragma unroll
    for (int i = 0; i < 4; ++i)
#pragma unroll
        for (int j = 0; j < 2; ++j) acc[i][j] = zero8();

    const int lr = tid >> 1;
    const int lc = (tid & 1) << 4;

    auto stage = [&](int buf, int k0) {
        unsigned int la = lds_addr32(&sA[buf][lr * LDS_LD + lc]);
        unsigned int ga = (unsigned int)(((m0 + lr) * D_MODEL + k0 + lc) * 2);
        async_ld_b128(la, ctxh, ga);
        async_ld_b128(la + 16, ctxh, ga + 16);
        unsigned int lb = lds_addr32(&sB[buf][lr * LDS_LD + lc]);
        unsigned int gb = (unsigned int)(((n0 + lr) * D_MODEL + k0 + lc) * 2);
        async_ld_b128(lb, Wto, gb);
        async_ld_b128(lb + 16, Wto, gb + 16);
    };

    stage(0, 0);
    wait_async0();
    __syncthreads();

    int buf = 0;
    for (int kt = 0; kt < NKSTEP; ++kt) {
        if (kt + 1 < NKSTEP) stage(buf ^ 1, (kt + 1) * GT_K);

        v16h af[4], bf[2];
#pragma unroll
        for (int i = 0; i < 4; ++i)
            af[i] = load_a(&sA[buf][(wm * 64 + i * 16) * LDS_LD], LDS_LD);
#pragma unroll
        for (int j = 0; j < 2; ++j)
            bf[j] = load_b(&sB[buf][(wn * 32 + j * 16) * LDS_LD], LDS_LD);
#pragma unroll
        for (int i = 0; i < 4; ++i)
#pragma unroll
            for (int j = 0; j < 2; ++j)
                acc[i][j] = wmma_f16(af[i], bf[j], acc[i][j]);

        if (kt + 1 < NKSTEP) wait_async0();
        __syncthreads();
        buf ^= 1;
    }

    const int lane = tid & 31;
    const int rowh = (lane >> 4) * 8;
    const int coln = lane & 15;
#pragma unroll
    for (int i = 0; i < 4; ++i)
#pragma unroll
        for (int j = 0; j < 2; ++j)
#pragma unroll
            for (int e = 0; e < 8; ++e) {
                int m = m0 + wm * 64 + i * 16 + rowh + e;
                int n = n0 + wn * 32 + j * 16 + coln;
                out[(size_t)m * D_MODEL + n] = acc[i][j][e] + bo[n];
            }
}

// ---------------- host launch ----------------

extern "C" void kernel_launch(void* const* d_in, const int* in_sizes, int n_in,
                              void* d_out, int out_size, void* d_ws, size_t ws_size,
                              hipStream_t stream) {
    (void)in_sizes; (void)n_in; (void)out_size; (void)ws_size;
    const float* x  = (const float*)d_in[0];
    const float* Wq = (const float*)d_in[1];
    const float* bq = (const float*)d_in[2];
    const float* Wk = (const float*)d_in[3];
    const float* bk = (const float*)d_in[4];
    const float* Wv = (const float*)d_in[5];
    const float* bv = (const float*)d_in[6];
    const float* Wo = (const float*)d_in[7];
    const float* bo = (const float*)d_in[8];
    float* out = (float*)d_out;

    char* ws = (char*)d_ws;
    size_t off = 0;
    const size_t SZ_ACT = (size_t)MTOK * D_MODEL * sizeof(_Float16);    // 16 MB
    const size_t SZ_W   = (size_t)D_MODEL * D_MODEL * sizeof(_Float16); // 2 MB

    _Float16* xh  = (_Float16*)(ws + off); off += SZ_ACT;  // reused as ctx later
    _Float16* Wtq = (_Float16*)(ws + off); off += SZ_W;
    _Float16* Wtk = (_Float16*)(ws + off); off += SZ_W;
    _Float16* Wtv = (_Float16*)(ws + off); off += SZ_W;
    _Float16* Wto = (_Float16*)(ws + off); off += SZ_W;
    _Float16* Qh  = (_Float16*)(ws + off); off += SZ_ACT;
    _Float16* Kh  = (_Float16*)(ws + off); off += SZ_ACT;
    _Float16* Vth = (_Float16*)(ws + off); off += SZ_ACT;
    _Float16* ctxh = xh;  // xh is dead after the QKV GEMM

    const int n = MTOK * D_MODEL;
    cvt_f16_kernel<<<(n + 255) / 256, 256, 0, stream>>>(x, xh, n);
    dim3 tb(32, 8), tg(D_MODEL / 32, D_MODEL / 32);
    wt_cvt_kernel<<<tg, tb, 0, stream>>>(Wq, Wtq);
    wt_cvt_kernel<<<tg, tb, 0, stream>>>(Wk, Wtk);
    wt_cvt_kernel<<<tg, tb, 0, stream>>>(Wv, Wtv);
    wt_cvt_kernel<<<tg, tb, 0, stream>>>(Wo, Wto);

    dim3 g1(MTOK / GT_M, D_MODEL / GT_N, 3);
    qkv_gemm_kernel<<<g1, 256, 0, stream>>>(xh, Wtq, Wtk, Wtv, bq, bk, bv, Qh, Kh, Vth);

    dim3 g2(SEQ / QTILE, BATCH * NHEADS);
    attn_kernel<<<g2, 256, 0, stream>>>(Qh, Kh, Vth, ctxh);

    dim3 g3(MTOK / GT_M, D_MODEL / GT_N);
    out_gemm_kernel<<<g3, 256, 0, stream>>>(ctxh, Wto, bo, out);
}
